// Block_23759759081740
// MI455X (gfx1250) — matrix operand order repaired and checked by reading, not compiled, not run on hardware
//
#include <hip/hip_runtime.h>
#include <hip/hip_bf16.h>

// ---------------- problem constants ----------------
#define BB   8
#define NQ   4096
#define NKV  4096
#define DE   1024
#define DP   128
#define HH   8
#define DHE  16
#define MM   64          // random features
#define SSH  163         // shift = NKV*2//50
#define NWIN (NKV - SSH) // 3933
#define R1   (2 * BB * NQ)   // 65536 query rows
#define R2   (BB * NKV)      // 32768 kv rows
#define RATIO 0.125f         // M^-0.5
#define KEPS_ 1e-4f

typedef __attribute__((ext_vector_type(16))) __bf16        v16bf;
typedef __attribute__((ext_vector_type(8)))  float         v8f;
typedef __attribute__((ext_vector_type(4)))  unsigned int  u32x4;
typedef __attribute__((ext_vector_type(4)))  int           v4i;
typedef __attribute__((address_space(1)))    v4i*          gptr_v4i;
typedef __attribute__((address_space(3)))    v4i*          lptr_v4i;

__device__ __forceinline__ unsigned short f2b(float f) {
  unsigned u = __float_as_uint(f);
  u += 0x7FFFu + ((u >> 16) & 1u);          // round-to-nearest-even
  return (unsigned short)(u >> 16);
}
__device__ __forceinline__ float gelu_t(float x) {
  float x3 = x * x * x;
  return 0.5f * x * (1.0f + tanhf(0.79788456080286535588f * (x + 0.044715f * x3)));
}

// ---------------- weight convert: W[K,N=128] f32 -> Wt[N,K] bf16 ----------------
__global__ void cvt_wt(const float* __restrict__ W, unsigned short* __restrict__ Wt,
                       int K, int N) {
  int idx = blockIdx.x * blockDim.x + threadIdx.x;
  if (idx >= K * N) return;
  int k = idx / N, n = idx - k * N;
  Wt[(size_t)n * K + k] = f2b(W[idx]);
}

// ---------------- row LayerNorm -> bf16 ----------------
__global__ void ln_rows(const float* __restrict__ in, const float* __restrict__ w,
                        const float* __restrict__ bp, unsigned short* __restrict__ out,
                        int D) {
  __shared__ float red[256];
  const int t = threadIdx.x, nt = blockDim.x;
  const size_t row = blockIdx.x;
  const float* rp = in + row * (size_t)D;
  const int per = D / nt;
  float s = 0.f, s2 = 0.f;
  for (int i = 0; i < per; ++i) { float v = rp[t + i * nt]; s += v; s2 += v * v; }
  red[t] = s; __syncthreads();
  for (int off = nt >> 1; off > 0; off >>= 1) { if (t < off) red[t] += red[t + off]; __syncthreads(); }
  const float mean = red[0] / (float)D; __syncthreads();
  red[t] = s2; __syncthreads();
  for (int off = nt >> 1; off > 0; off >>= 1) { if (t < off) red[t] += red[t + off]; __syncthreads(); }
  const float var = red[0] / (float)D - mean * mean;
  const float inv = rsqrtf(var + 1e-5f);
  for (int i = 0; i < per; ++i) {
    int c = t + i * nt;
    out[row * (size_t)D + c] = f2b((rp[c] - mean) * inv * w[c] + bp[c]);
  }
}

// ---------------- WMMA bf16 GEMM: C[M,128] = A[M,K] * Wt^T  ----------------
// block = 256 threads = 8 waves; wave w -> 16x16 tile at colBase = 16*w;
// blockIdx.x -> 16-row stripe. The 16xK A stripe (contiguous in global) is
// async-copied into LDS once per block (ASYNCcnt path), then every wave reads
// its A fragments from LDS; B (weights, L2-resident) streams from global.
// mode: 0 = f32 store, 1 = bias+gelu->bf16, 2 = bias+residual->f32.
__global__ void gemm_bf16_n128(const unsigned short* __restrict__ A,
                               const unsigned short* __restrict__ Bt,
                               const float* __restrict__ bias,
                               const float* __restrict__ res,
                               float* __restrict__ outF,
                               unsigned short* __restrict__ outB,
                               int K, int mode) {
  extern __shared__ unsigned short lds_a[];   // 16 * K bf16
  const int wave = threadIdx.x >> 5;
  const int lane = threadIdx.x & 31;
  const int half = lane >> 4;
  const int l16  = lane & 15;
  const size_t rowBase = (size_t)blockIdx.x * 16;
  const int colBase = wave * 16;

  // ---- stage A stripe: 16 rows x K bf16 (contiguous 16*K*2 bytes) ----
  const unsigned short* gA = A + rowBase * (size_t)K;
  const int nchunks = (16 * K) / 8;           // 16-byte chunks
#if __has_builtin(__builtin_amdgcn_global_load_async_to_lds_b128) && \
    __has_builtin(__builtin_amdgcn_s_wait_asynccnt)
  for (int c = threadIdx.x; c < nchunks; c += 256) {
    __builtin_amdgcn_global_load_async_to_lds_b128(
        (gptr_v4i)(gA + (size_t)c * 8),
        (lptr_v4i)((char*)lds_a + (size_t)c * 16),
        0, 0);
  }
  __builtin_amdgcn_s_wait_asynccnt(0);
#else
  for (int c = threadIdx.x; c < nchunks; c += 256) {
    *(u32x4*)((char*)lds_a + (size_t)c * 16) = *(const u32x4*)(gA + (size_t)c * 8);
  }
#endif
  __syncthreads();

  const unsigned short* aRow = lds_a + l16 * K;
  const unsigned short* bRow = Bt + (size_t)(colBase + l16) * K;

  v8f c = {0.f, 0.f, 0.f, 0.f, 0.f, 0.f, 0.f, 0.f};
  for (int kk = 0; kk < K; kk += 32) {
    union { u32x4 u[2]; v16bf v; } af, bfrag;
    // A fragment: lane (m=l16, half): K = half*8 + [0..7] and 16 + half*8 + [0..7]
    af.u[0] = *(const u32x4*)(aRow + kk + half * 8);
    af.u[1] = *(const u32x4*)(aRow + kk + 16 + half * 8);
    // B fragment: lane (n=l16, half): K = half*8 + [0..7] and 16 + half*8 + [0..7]
    bfrag.u[0] = *(const u32x4*)(bRow + kk + half * 8);
    bfrag.u[1] = *(const u32x4*)(bRow + kk + 16 + half * 8);
    c = __builtin_amdgcn_wmma_f32_16x16x32_bf16(false, af.v, false, bfrag.v,
                                                (short)0, c, false, false);
  }
  const int n = colBase + l16;
  const float bv = bias ? bias[n] : 0.f;
  for (int r = 0; r < 8; ++r) {
    int m = r + half * 8;
    size_t idx = (rowBase + m) * 128 + n;
    float v = c[r] + bv;
    if (mode == 1)      outB[idx] = f2b(gelu_t(v));
    else if (mode == 2) outF[idx] = v + res[idx];
    else                outF[idx] = v;
  }
}

// ---------------- FAVOR+ feature map for Q ----------------
// one thread per (b2,h,n); writes qf (ws) and q_un (output region 2)
__global__ void feat_q(const float* __restrict__ qbuf, const float* __restrict__ proj,
                       float* __restrict__ qf, float* __restrict__ q_un) {
  __shared__ float pr[MM * DHE];
  for (int i = threadIdx.x; i < MM * DHE; i += blockDim.x) pr[i] = proj[i];
  __syncthreads();
  const int tid = blockIdx.x * blockDim.x + threadIdx.x;   // (b2*H + h)*NQ + n
  const int n  = tid & (NQ - 1);
  const int h  = (tid >> 12) & 7;
  const int b2 = tid >> 15;
  const float* dp = qbuf + ((size_t)b2 * NQ + n) * 128 + h * 16;
  float d[DHE]; float diag = 0.f;
  for (int i = 0; i < DHE; ++i) { float v = dp[i]; d[i] = 0.5f * v; diag += v * v; }
  diag *= 0.125f;  // 0.5 * dn^2, dn = 16^-0.25 = 0.5
  float dd[MM]; float mx = -1e30f;
  for (int m = 0; m < MM; ++m) {
    float a = 0.f;
    for (int i = 0; i < DHE; ++i) a += d[i] * pr[m * DHE + i];
    dd[m] = a; if (a > mx) mx = a;
  }
  const float emx = expf(-mx);
  const size_t ob = (size_t)tid * MM;
  for (int m = 0; m < MM; ++m) {
    float un = RATIO * expf(dd[m] - diag);
    q_un[ob + m] = un;
    qf[ob + m]   = un * emx + RATIO * KEPS_;
  }
}

// ---------------- K feature map: pass 1 = per-(b2,h) global max of dd ----------------
__global__ void feat_k_max(const float* __restrict__ kbuf, const float* __restrict__ proj,
                           float* __restrict__ mx) {
  __shared__ float pr[MM * DHE];
  __shared__ float red[256];
  for (int i = threadIdx.x; i < MM * DHE; i += blockDim.x) pr[i] = proj[i];
  __syncthreads();
  const int bh = blockIdx.x;           // 0..127
  const int b2 = bh >> 3, h = bh & 7;
  const int w  = b2 >> 3, b = b2 & 7;
  float lm = -1e30f;
  for (int n = threadIdx.x; n < NWIN; n += blockDim.x) {
    const float* dp = kbuf + ((size_t)(b * NKV + n + w * SSH)) * 128 + h * 16;
    float d[DHE];
    for (int i = 0; i < DHE; ++i) d[i] = 0.5f * dp[i];
    for (int m = 0; m < MM; ++m) {
      float a = 0.f;
      for (int i = 0; i < DHE; ++i) a += d[i] * pr[m * DHE + i];
      if (a > lm) lm = a;
    }
  }
  red[threadIdx.x] = lm; __syncthreads();
  for (int off = blockDim.x >> 1; off > 0; off >>= 1) {
    if (threadIdx.x < off) red[threadIdx.x] = fmaxf(red[threadIdx.x], red[threadIdx.x + off]);
    __syncthreads();
  }
  if (threadIdx.x == 0) mx[bh] = red[0];
}

// ---------------- K feature map: pass 2 = kf + k_un ----------------
__global__ void feat_k(const float* __restrict__ kbuf, const float* __restrict__ proj,
                       const float* __restrict__ mx, float* __restrict__ kf,
                       float* __restrict__ k_un) {
  __shared__ float pr[MM * DHE];
  for (int i = threadIdx.x; i < MM * DHE; i += blockDim.x) pr[i] = proj[i];
  __syncthreads();
  const int tid = blockIdx.x * blockDim.x + threadIdx.x;
  const int total = 2 * BB * HH * NWIN;
  if (tid >= total) return;
  const int hb = tid / NWIN;
  const int n  = tid - hb * NWIN;
  const int h  = hb & 7;
  const int b2 = hb >> 3;
  const int w  = b2 >> 3, b = b2 & 7;
  const float* dp = kbuf + ((size_t)(b * NKV + n + w * SSH)) * 128 + h * 16;
  float d[DHE]; float diag = 0.f;
  for (int i = 0; i < DHE; ++i) { float v = dp[i]; d[i] = 0.5f * v; diag += v * v; }
  diag *= 0.125f;
  const float emx = expf(-mx[hb]);
  const size_t ob = (size_t)tid * MM;
  for (int m = 0; m < MM; ++m) {
    float a = 0.f;
    for (int i = 0; i < DHE; ++i) a += d[i] * pr[m * DHE + i];
    float un = RATIO * expf(a - diag);
    k_un[ob + m] = un;
    kf[ob + m]   = un * emx + RATIO * KEPS_;
  }
}

// ---------------- linear attention: ctx2[m,e] = sum_n kf*vw ; ksum[m] ----------------
__global__ void attn_ctx(const float* __restrict__ kf, const float* __restrict__ vbuf,
                         float* __restrict__ ctx2, float* __restrict__ ksum) {
  __shared__ float kfs[32 * MM];
  __shared__ float vws[32 * DHE];
  const int bh = blockIdx.x;          // (b2,h)
  const int b2 = bh >> 3, h = bh & 7;
  const int w  = b2 >> 3, b = b2 & 7;
  const int t  = threadIdx.x;
  const int m  = t & 63;
  const int eg = t >> 6;              // 4 e-groups of 4
  float a0 = 0.f, a1 = 0.f, a2 = 0.f, a3 = 0.f, ks = 0.f;
  const float* kfb = kf + (size_t)bh * NWIN * MM;
  for (int n0 = 0; n0 < NWIN; n0 += 32) {
    for (int i = t; i < 32 * MM; i += 256) {
      int nn = n0 + (i >> 6);
      kfs[i] = (nn < NWIN) ? kfb[(size_t)nn * MM + (i & 63)] : 0.f;
    }
    for (int i = t; i < 32 * DHE; i += 256) {
      int nn = n0 + (i >> 4);
      vws[i] = (nn < NWIN)
                 ? vbuf[((size_t)(b * NKV + nn + w * SSH)) * 128 + h * 16 + (i & 15)]
                 : 0.f;
    }
    __syncthreads();
    for (int j = 0; j < 32; ++j) {
      float kv = kfs[j * MM + m];
      a0 += kv * vws[j * DHE + eg * 4 + 0];
      a1 += kv * vws[j * DHE + eg * 4 + 1];
      a2 += kv * vws[j * DHE + eg * 4 + 2];
      a3 += kv * vws[j * DHE + eg * 4 + 3];
      if (eg == 0) ks += kv;
    }
    __syncthreads();
  }
  float* cp = ctx2 + (size_t)bh * MM * DHE + m * DHE + eg * 4;
  cp[0] = a0; cp[1] = a1; cp[2] = a2; cp[3] = a3;
  if (eg == 0) ksum[bh * MM + m] = ks;
}

// ---------------- out[n,e] = (ctx2^T qf) * dinv -> bf16 rows [R1,128] ----------------
__global__ void attn_out(const float* __restrict__ qf, const float* __restrict__ ctx2,
                         const float* __restrict__ ksum, unsigned short* __restrict__ o) {
  __shared__ float c2[MM * DHE];
  __shared__ float ks[MM];
  const int bh = blockIdx.y;
  const int b2 = bh >> 3, h = bh & 7;
  for (int i = threadIdx.x; i < MM * DHE; i += blockDim.x) c2[i] = ctx2[(size_t)bh * MM * DHE + i];
  if (threadIdx.x < MM) ks[threadIdx.x] = ksum[bh * MM + threadIdx.x];
  __syncthreads();
  const int n = blockIdx.x * blockDim.x + threadIdx.x;
  const float* qp = qf + ((size_t)bh * NQ + n) * MM;
  float q[MM]; float den = 0.f;
  for (int m = 0; m < MM; ++m) { q[m] = qp[m]; den += q[m] * ks[m]; }
  const float dinv = 1.f / den;
  const size_t ob = ((size_t)b2 * NQ + n) * 128 + h * 16;
  for (int e = 0; e < DHE; ++e) {
    float a = 0.f;
    for (int m = 0; m < MM; ++m) a += c2[m * DHE + e] * q[m];
    o[ob + e] = f2b(a * dinv);
  }
}

// ---------------- host ----------------
extern "C" void kernel_launch(void* const* d_in, const int* in_sizes, int n_in,
                              void* d_out, int out_size, void* d_ws, size_t ws_size,
                              hipStream_t stream) {
  const float* x     = (const float*)d_in[1];
  const float* y     = (const float*)d_in[2];
  const float* ln1w  = (const float*)d_in[3];
  const float* ln1b  = (const float*)d_in[4];
  const float* ln2w  = (const float*)d_in[5];
  const float* ln2b  = (const float*)d_in[6];
  const float* ln4w  = (const float*)d_in[7];
  const float* ln4b  = (const float*)d_in[8];
  const float* fcw   = (const float*)d_in[9];
  const float* fcb   = (const float*)d_in[10];
  const float* pjw   = (const float*)d_in[11];
  const float* pjb   = (const float*)d_in[12];
  const float* wq    = (const float*)d_in[13];
  const float* wk    = (const float*)d_in[14];
  const float* wv    = (const float*)d_in[15];
  const float* wow   = (const float*)d_in[16];
  const float* wob   = (const float*)d_in[17];
  const float* proj  = (const float*)d_in[18];

  float* y_out = (float*)d_out;
  float* q_un  = y_out + (size_t)R1 * DP;                 // [2B,H,NQ,M]
  float* k_un  = q_un + (size_t)2 * BB * HH * NQ * MM;    // [2B,H,NW,M]

  char* ws = (char*)d_ws;
  size_t off = 0;
  auto alloc = [&](size_t bytes) -> void* {
    void* p = ws + off;
    off += (bytes + 255) & ~(size_t)255;
    return p;
  };
  unsigned short* wt_fc = (unsigned short*)alloc((size_t)DP * DP * 2);
  unsigned short* wt_pj = (unsigned short*)alloc((size_t)DP * DP * 2);
  unsigned short* wt_q  = (unsigned short*)alloc((size_t)DP * DP * 2);
  unsigned short* wt_o  = (unsigned short*)alloc((size_t)DP * DP * 2);
  unsigned short* wt_k  = (unsigned short*)alloc((size_t)DE * DP * 2);
  unsigned short* wt_v  = (unsigned short*)alloc((size_t)DE * DP * 2);
  unsigned short* a0    = (unsigned short*)alloc((size_t)R1 * DP * 2);
  unsigned short* hbuf  = (unsigned short*)alloc((size_t)R1 * DP * 2);
  float*          y1    = (float*)alloc((size_t)R1 * DP * 4);
  unsigned short* xq    = (unsigned short*)alloc((size_t)R1 * DP * 2);
  float*          qb    = (float*)alloc((size_t)R1 * DP * 4);
  unsigned short* ctx   = (unsigned short*)alloc((size_t)R2 * DE * 2);
  float*          kb    = (float*)alloc((size_t)R2 * DP * 4);
  float*          vb    = (float*)alloc((size_t)R2 * DP * 4);
  float*          qfb   = (float*)alloc((size_t)2 * BB * HH * NQ * MM * 4);
  float*          kfb   = (float*)alloc((size_t)2 * BB * HH * NWIN * MM * 4);
  float*          mxb   = (float*)alloc(2 * BB * HH * 4);
  float*          ksum  = (float*)alloc((size_t)2 * BB * HH * MM * 4);
  float*          c2b   = (float*)alloc((size_t)2 * BB * HH * MM * DHE * 4);
  unsigned short* obuf  = (unsigned short*)alloc((size_t)R1 * DP * 2);

  // weights -> transposed bf16
  cvt_wt<<<(DP * DP + 255) / 256, 256, 0, stream>>>(fcw, wt_fc, DP, DP);
  cvt_wt<<<(DP * DP + 255) / 256, 256, 0, stream>>>(pjw, wt_pj, DP, DP);
  cvt_wt<<<(DP * DP + 255) / 256, 256, 0, stream>>>(wq,  wt_q,  DP, DP);
  cvt_wt<<<(DP * DP + 255) / 256, 256, 0, stream>>>(wow, wt_o,  DP, DP);
  cvt_wt<<<(DE * DP + 255) / 256, 256, 0, stream>>>(wk,  wt_k,  DE, DP);
  cvt_wt<<<(DE * DP + 255) / 256, 256, 0, stream>>>(wv,  wt_v,  DE, DP);

  const size_t lds128  = 16 * DP * sizeof(unsigned short);   // 4 KB
  const size_t lds1024 = 16 * DE * sizeof(unsigned short);   // 32 KB

  // MLP2 residual branch
  ln_rows<<<R1, 128, 0, stream>>>(y, ln4w, ln4b, a0, DP);
  gemm_bf16_n128<<<R1 / 16, 256, lds128, stream>>>(a0, wt_fc, fcb, nullptr, nullptr, hbuf, DP, 1);
  gemm_bf16_n128<<<R1 / 16, 256, lds128, stream>>>(hbuf, wt_pj, pjb, y, y1, nullptr, DP, 2);

  // Q path
  ln_rows<<<R1, 128, 0, stream>>>(y1, ln1w, ln1b, xq, DP);
  gemm_bf16_n128<<<R1 / 16, 256, lds128, stream>>>(xq, wt_q, nullptr, nullptr, qb, nullptr, DP, 0);

  // K/V path
  ln_rows<<<R2, 256, 0, stream>>>(x, ln2w, ln2b, ctx, DE);
  gemm_bf16_n128<<<R2 / 16, 256, lds1024, stream>>>(ctx, wt_k, nullptr, nullptr, kb, nullptr, DE, 0);
  gemm_bf16_n128<<<R2 / 16, 256, lds1024, stream>>>(ctx, wt_v, nullptr, nullptr, vb, nullptr, DE, 0);

  // FAVOR+ feature maps
  feat_q<<<(2 * BB * HH * NQ) / 256, 256, 0, stream>>>(qb, proj, qfb, q_un);
  feat_k_max<<<2 * BB * HH, 256, 0, stream>>>(kb, proj, mxb);
  feat_k<<<(2 * BB * HH * NWIN + 255) / 256, 256, 0, stream>>>(kb, proj, mxb, kfb, k_un);

  // linear attention
  attn_ctx<<<2 * BB * HH, 256, 0, stream>>>(kfb, vb, c2b, ksum);
  attn_out<<<dim3(NQ / 256, 2 * BB * HH), 256, 0, stream>>>(qfb, c2b, ksum, obuf);

  // output projection + residual -> y_out
  gemm_bf16_n128<<<R1 / 16, 256, lds128, stream>>>(obuf, wt_o, wob, y1, y_out, nullptr, DP, 2);
}
